// DecisionStep_40106404610231
// MI455X (gfx1250) — matrix operand order repaired and checked by reading, not compiled, not run on hardware
//
#include <hip/hip_runtime.h>
#include <hip/hip_bf16.h>

// ---------------------------------------------------------------------------
// TabNet decision step, fused for MI455X (gfx1250, wave32, WMMA).
// grid = 512 workgroups (one per 128-row ghost-BN virtual batch), 256 threads.
// GEMMs in v_wmma_f32_16x16x32_bf16 with f32 accumulation; BN / sparsemax /
// GLU / residual all workgroup-local so HBM is touched once per tensor.
// ---------------------------------------------------------------------------

typedef __bf16 bf16_t;
typedef __attribute__((ext_vector_type(16))) __bf16 v16bf;
typedef __attribute__((ext_vector_type(8)))  __bf16 v8bf;
typedef __attribute__((ext_vector_type(4)))  __bf16 v4bf;
typedef __attribute__((ext_vector_type(8)))  float  v8f;
typedef __attribute__((ext_vector_type(4)))  float  v4f;

#define B_ROWS   65536
#define DIM      256      // INPUT_DIM == N_OUT
#define NA       128
#define H2       512      // 2*N_OUT
#define BV       128      // ghost-BN virtual batch == rows per workgroup
#define A_S      264      // bf16 LDS stride for A staging (16B aligned, bank-skewed)
#define H_S      520      // bf16 LDS stride for h staging
#define BLOCK    256
#define NWGS     (B_ROWS / BV)
// LDS: stats(1024 f32) + loss(4 f32) + sA(128*A_S bf16) + sH(128*H_S bf16)
#define LDS_BYTES (4096 + 16 + BV * A_S * 2 + BV * H_S * 2)

// Native converts on gfx1250 (single v_cvt instead of bit-twiddling).
static __device__ inline bf16_t f2bf(float f)  { return (bf16_t)f; }
static __device__ inline float  bf2f(bf16_t b) { return (float)b; }

static __device__ inline float wred_sum(float v) {
#pragma unroll
  for (int off = 16; off > 0; off >>= 1) v += __shfl_xor(v, off, 32);
  return v;
}
static __device__ inline float wred_max(float v) {
#pragma unroll
  for (int off = 16; off > 0; off >>= 1) v = fmaxf(v, __shfl_xor(v, off, 32));
  return v;
}

// A fragment (16x32 bf16, M x K). ISA layout: lane m and lane m+16 hold row m;
// lanes 0-15 carry K = k0 + {0..7, 16..23}, lanes 16-31 K = k0 + {8..15, 24..31}.
// Each half is 8 contiguous bf16 -> two 16B LDS loads.
static __device__ inline v16bf load_frag_A(const bf16_t* base, int stride,
                                           int mrow, int k0, int lane) {
  int half = lane >> 4;
  int m    = lane & 15;
  const bf16_t* p = base + (mrow + m) * stride + k0 + half * 8;
  v8bf lo = *(const v8bf*)p;          // K = k0 + half*8 + 0..7      (VGPR 0..3)
  v8bf hi = *(const v8bf*)(p + 16);   // K = k0 + 16 + half*8 + 0..7 (VGPR 4..7)
  return __builtin_shufflevector(lo, hi, 0, 1, 2, 3, 4, 5, 6, 7,
                                 8, 9, 10, 11, 12, 13, 14, 15);
}

// B fragment (32x16 bf16, K x N) read straight from row-major W[n][k] (== B^T).
// Lane holds column n = n0 + (lane&15); K = kb + half*8 + {0..7, 16..23}.
static __device__ inline v16bf load_frag_B(const bf16_t* __restrict__ W, int Ktot,
                                           int n0, int kb, int lane) {
  int half = lane >> 4;
  int n    = n0 + (lane & 15);
  const bf16_t* p = W + n * Ktot + kb + half * 8;
  v8bf lo = *(const v8bf*)p;
  v8bf hi = *(const v8bf*)(p + 16);
  return __builtin_shufflevector(lo, hi, 0, 1, 2, 3, 4, 5, 6, 7,
                                 8, 9, 10, 11, 12, 13, 14, 15);
}

// D tile (16x16 f32): lane<16 -> (m = r, n = lane); lane>=16 -> (m = r+8).
// Store h as bf16 to LDS and push per-column sum/sumsq partials (ds_add_f32).
static __device__ inline void store_tile_stats(v8f acc, bf16_t* sH, int hs,
                                               int mrow, int n0, int lane,
                                               float* ssum, float* ssq) {
  int n = n0 + (lane & 15);
  int half = lane >> 4;
  float s = 0.f, q = 0.f;
#pragma unroll
  for (int r = 0; r < 8; ++r) {
    float v = acc[r];
    s += v;
    q += v * v;
    sH[(mrow + half * 8 + r) * hs + n] = f2bf(v);
  }
  atomicAdd(&ssum[n], s);
  atomicAdd(&ssq[n], q);
}

// f32 -> bf16 weight pre-conversion into d_ws; also zeroes the loss slot.
__global__ void convert_weights_kernel(const float* __restrict__ Wa,
                                       const float* __restrict__ W0,
                                       const float* __restrict__ W1,
                                       bf16_t* __restrict__ ws,
                                       float* __restrict__ loss_out) {
  int i = blockIdx.x * blockDim.x + threadIdx.x;
  if (i == 0) *loss_out = 0.f;
  const int nWa = DIM * NA;          // 32768
  const int nW0 = H2 * DIM;          // 131072
  if (i < nWa)                 ws[i] = f2bf(Wa[i]);
  else if (i < nWa + nW0)      ws[i] = f2bf(W0[i - nWa]);
  else if (i < nWa + 2 * nW0)  ws[i] = f2bf(W1[i - nWa - nW0]);
}

__global__ __launch_bounds__(BLOCK)
void tabnet_step_kernel(const float* __restrict__ x,
                        const float* __restrict__ a,
                        const float* __restrict__ priors,
                        const bf16_t* __restrict__ Wa,
                        const float* __restrict__ g_a, const float* __restrict__ bt_a,
                        const bf16_t* __restrict__ W0,
                        const float* __restrict__ g0, const float* __restrict__ bt0,
                        const bf16_t* __restrict__ W1,
                        const float* __restrict__ g1, const float* __restrict__ bt1,
                        float* __restrict__ out, float* __restrict__ loss_out) {
  extern __shared__ char smem_raw[];
  float*  stats_mu = (float*)smem_raw;       // 512: col sums, then means
  float*  stats_rs = stats_mu + 512;         // 512: col sumsq, then rsqrt(var+eps)
  float*  lossAcc  = stats_rs + 512;         // [0] loss accumulator
  bf16_t* sA = (bf16_t*)(lossAcc + 4);       // A staging  [128][A_S]
  bf16_t* sH = sA + BV * A_S;                // h staging  [128][H_S]

  const int tid  = threadIdx.x;
  const int wave = tid >> 5;
  const int lane = tid & 31;
  const long rowBase = (long)blockIdx.x * BV;

  // ---- Phase 0: stage `a` block (f32 -> bf16), zero stats & loss ------------
  // b128 global loads (4 floats) + 8B LDS stores (4 bf16) per iteration.
  {
    const v4f* a4 = (const v4f*)(a + rowBase * NA);
    for (int i = tid; i < BV * NA / 4; i += BLOCK) {
      int m  = i >> 5;             // 32 float4 per 128-wide row
      int k4 = (i & 31) * 4;
      v4f v = a4[i];
      v4bf b = { f2bf(v.x), f2bf(v.y), f2bf(v.z), f2bf(v.w) };
      *(v4bf*)(sA + m * A_S + k4) = b;
    }
  }
  for (int i = tid; i < 1024; i += BLOCK) stats_mu[i] = 0.f;
  if (tid == 0) lossAcc[0] = 0.f;
  __syncthreads();

  // ---- Phase 1: attentive GEMM  h = a @ Wa^T  (128x256, K=128) --------------
  // 16 column tiles; wave owns 2 tiles, caches B frags in VGPRs, sweeps 8 row strips.
  for (int tt = 0; tt < 2; ++tt) {
    int n0 = (wave * 2 + tt) * 16;
    v16bf bfr[4];
#pragma unroll
    for (int kb = 0; kb < 4; ++kb) bfr[kb] = load_frag_B(Wa, NA, n0, kb * 32, lane);
    for (int s = 0; s < 8; ++s) {
      v8f acc = {};
#pragma unroll
      for (int kb = 0; kb < 4; ++kb) {
        v16bf afr = load_frag_A(sA, A_S, s * 16, kb * 32, lane);
        acc = __builtin_amdgcn_wmma_f32_16x16x32_bf16(false, afr, false, bfr[kb],
                                                      (short)0, acc, false, false);
      }
      store_tile_stats(acc, sH, H_S, s * 16, n0, lane, stats_mu, stats_rs);
    }
  }
  __syncthreads();

  // ---- Phase 2: ghost-BN stats for the 256 attentive columns ----------------
  {
    float mu  = stats_mu[tid] * (1.f / 128.f);
    float var = stats_rs[tid] * (1.f / 128.f) - mu * mu;
    stats_mu[tid] = mu;
    stats_rs[tid] = rsqrtf(var + 1e-5f);
  }
  __syncthreads();

  // ---- Phase 3: sparsemax per row (bisection on tau), z = x*mask -> sA ------
  float lossLocal = 0.f;
  for (int rr = 0; rr < 16; ++rr) {
    int m = wave * 16 + rr;
    long rowG = rowBase + m;
    float v[8];
    float vmax = -3.4e38f;
#pragma unroll
    for (int j = 0; j < 8; ++j) {
      int c = lane + j * 32;
      float h = bf2f(sH[m * H_S + c]);
      float nrm = (h - stats_mu[c]) * stats_rs[c] * g_a[c] + bt_a[c];
      float z = nrm * priors[rowG * DIM + c];
      v[j] = z;
      vmax = fmaxf(vmax, z);
    }
    vmax = wred_max(vmax);
    float lo = vmax - 1.f, hi = vmax;
    for (int it = 0; it < 26; ++it) {       // root of sum(relu(z - tau)) == 1
      float tau = 0.5f * (lo + hi);
      float ssum = 0.f;
#pragma unroll
      for (int j = 0; j < 8; ++j) ssum += fmaxf(v[j] - tau, 0.f);
      ssum = wred_sum(ssum);
      if (ssum > 1.f) lo = tau; else hi = tau;
    }
    float tau = 0.5f * (lo + hi);
#pragma unroll
    for (int j = 0; j < 8; ++j) {
      int c = lane + j * 32;
      float mk = fmaxf(v[j] - tau, 0.f);
      lossLocal += -mk * __logf(mk + 1e-10f);
      sA[m * A_S + c] = f2bf(x[rowG * DIM + c] * mk);
    }
  }
  atomicAdd(lossAcc, lossLocal);
  __syncthreads();
  for (int i = tid; i < 1024; i += BLOCK) stats_mu[i] = 0.f;
  __syncthreads();

  // ---- Two GLU blocks: GEMM -> ghost-BN -> GLU -> residual ------------------
  const bf16_t* Wl[2] = { W0, W1 };
  const float*  gl[2] = { g0, g1 };
  const float*  bl[2] = { bt0, bt1 };

  for (int layer = 0; layer < 2; ++layer) {
    // GEMM: h = z @ W^T (128x512, K=256). 32 col tiles; wave owns 4.
    for (int tt = 0; tt < 4; ++tt) {
      int n0 = (wave * 4 + tt) * 16;
      // Prefetch next tile's weights while this tile computes.
      if (tt < 3)
        __builtin_prefetch(Wl[layer] + (n0 + 16) * DIM + lane * 16, 0, 0);
      v16bf bfr[8];
#pragma unroll
      for (int kb = 0; kb < 8; ++kb)
        bfr[kb] = load_frag_B(Wl[layer], DIM, n0, kb * 32, lane);
      for (int s = 0; s < 8; ++s) {
        v8f acc = {};
#pragma unroll
        for (int kb = 0; kb < 8; ++kb) {
          v16bf afr = load_frag_A(sA, A_S, s * 16, kb * 32, lane);
          acc = __builtin_amdgcn_wmma_f32_16x16x32_bf16(false, afr, false, bfr[kb],
                                                        (short)0, acc, false, false);
        }
        store_tile_stats(acc, sH, H_S, s * 16, n0, lane, stats_mu, stats_rs);
      }
    }
    __syncthreads();

    // Ghost-BN stats for all 512 columns.
    for (int c = tid; c < 512; c += BLOCK) {
      float mu  = stats_mu[c] * (1.f / 128.f);
      float var = stats_rs[c] * (1.f / 128.f) - mu * mu;
      stats_mu[c] = mu;
      stats_rs[c] = rsqrtf(var + 1e-5f);
    }
    __syncthreads();

    // GLU + residual: thread t owns column c = t for all 128 rows.
    {
      int c = tid;
      float mu1 = stats_mu[c],        rs1 = stats_rs[c];
      float ga1 = gl[layer][c],       be1 = bl[layer][c];
      float mu2 = stats_mu[c + 256],  rs2 = stats_rs[c + 256];
      float ga2 = gl[layer][c + 256], be2 = bl[layer][c + 256];
      for (int m = 0; m < BV; ++m) {
        float h1 = (bf2f(sH[m * H_S + c])       - mu1) * rs1 * ga1 + be1;
        float h2 = (bf2f(sH[m * H_S + c + 256]) - mu2) * rs2 * ga2 + be2;
        float glu = h1 * (1.f / (1.f + __expf(-h2)));
        float zn = (bf2f(sA[m * A_S + c]) + glu) * 0.70710678f;
        if (layer == 0) {
          sA[m * A_S + c] = f2bf(zn);
        } else {
          out[(rowBase + m) * DIM + c] = zn;
        }
      }
    }
    __syncthreads();
    if (layer == 0) {
      for (int i = tid; i < 1024; i += BLOCK) stats_mu[i] = 0.f;
      __syncthreads();
    }
  }

  // ---- Sparse loss: mean over B*DIM ----------------------------------------
  if (tid == 0)
    atomicAdd(loss_out, lossAcc[0] * (1.f / ((float)B_ROWS * (float)DIM)));
}

extern "C" void kernel_launch(void* const* d_in, const int* in_sizes, int n_in,
                              void* d_out, int out_size, void* d_ws, size_t ws_size,
                              hipStream_t stream) {
  (void)in_sizes; (void)n_in; (void)out_size; (void)ws_size;
  const float* x      = (const float*)d_in[0];
  const float* a      = (const float*)d_in[1];
  const float* priors = (const float*)d_in[2];
  const float* W_a    = (const float*)d_in[3];
  // d_in[4] (b_a), d_in[8] (b0), d_in[12] (b1): pre-BN biases cancel in GhostBN.
  const float* g_a    = (const float*)d_in[5];
  const float* bt_a   = (const float*)d_in[6];
  const float* W0     = (const float*)d_in[7];
  const float* g0     = (const float*)d_in[9];
  const float* bt0    = (const float*)d_in[10];
  const float* W1     = (const float*)d_in[11];
  const float* g1     = (const float*)d_in[13];
  const float* bt1    = (const float*)d_in[14];

  float* out      = (float*)d_out;
  float* loss_out = out + (size_t)B_ROWS * DIM;   // tuple: (z, sparse_loss)

  bf16_t* wbf = (bf16_t*)d_ws;                    // [Wa | W0 | W1] in bf16
  const int nWa = DIM * NA, nW0 = H2 * DIM;
  const int ntot = nWa + 2 * nW0;

  convert_weights_kernel<<<(ntot + BLOCK - 1) / BLOCK, BLOCK, 0, stream>>>(
      W_a, W0, W1, wbf, loss_out);

  tabnet_step_kernel<<<NWGS, BLOCK, LDS_BYTES, stream>>>(
      x, a, priors,
      wbf, g_a, bt_a,
      wbf + nWa, g0, bt0,
      wbf + nWa + nW0, g1, bt1,
      out, loss_out);
}
